// MultiHeadAttention_89610197663897
// MI455X (gfx1250) — matrix operand order repaired and checked by reading, not compiled, not run on hardware
//
#include <hip/hip_runtime.h>

// ---------------------------------------------------------------------------
// MHA forward for B=2, T=2048, C=1024, H=16, d=64 on gfx1250 (wave32, WMMA).
// Matmuls via v_wmma_f32_16x16x32_f16; weight tiles staged in LDS by the
// Tensor Data Mover (tensor_load_to_lds, double-buffered, s_wait_tensorcnt).
// B fragments are batch-loaded into distinct registers so the 8 WMMAs per
// k-step issue back-to-back behind a single s_wait_dscnt.
// ---------------------------------------------------------------------------

typedef __attribute__((ext_vector_type(16))) _Float16 v16h;
typedef __attribute__((ext_vector_type(8)))  _Float16 v8h;
typedef __attribute__((ext_vector_type(8)))  float    v8f;
typedef __attribute__((ext_vector_type(4)))  unsigned int v4u;
typedef __attribute__((ext_vector_type(8)))  int v8i;
typedef __attribute__((ext_vector_type(4)))  int v4i;

#define CAT16(lo,hi) __builtin_shufflevector((lo),(hi),0,1,2,3,4,5,6,7,8,9,10,11,12,13,14,15)
#define WMMA_F16(a,b,c) __builtin_amdgcn_wmma_f32_16x16x32_f16(false,(a),false,(b),(short)0,(c),false,false)

#define Bn 2
#define Tn 2048
#define Cn 1024
#define Hn 16
#define Dn 64
#define BT (Bn*Tn)

// ---- TDM: DMA a 32-row x 32-f16 tile (row stride 1024 f16) into LDS -------
// D# per CDNA5 ISA ch.8: group0 = {count/type, lds_addr, global_addr},
// group1 = {data_size, tensor dims/strides, tile dims}. 2D tensor -> remaining
// groups zero. Tracked with TENSORcnt (in-order per wave).
// This toolchain's builtin is the 6-arg form:
//   (uint32x4 g0, int32x8 g1, int32x4, int32x4, int32x8, i32 cpol)
__device__ __forceinline__ void tdm_load_tile32x32(const _Float16* g, unsigned lds_byte_addr) {
  unsigned long long ga = (unsigned long long)(uintptr_t)g;
  v4u g0 = { 1u,                                   // count=1, is_restore=0, gather off
             lds_byte_addr,                        // lds_addr[31:0]
             (unsigned)ga,                         // global_addr[31:0]
             (unsigned)(ga >> 32) | 0x80000000u }; // global_addr[56:32] | type=2
  v8i g1 = { (int)(1u << 16),                      // workgroup_mask=0, data_size=1 (2B)
             (int)(1024u << 16),                   // tensor_dim0[15:0] = 1024
             (int)(1024u << 16),                   // tensor_dim0 hi=0 | tensor_dim1[15:0]=1024
             (int)(32u << 16),                     // tensor_dim1 hi=0 | tile_dim0=32
             32,                                   // tile_dim1=32, tile_dim2=0
             1024,                                 // tensor_dim0_stride[31:0] = 1024
             0, 0 };                               // stride hi / dim1_stride = 0
  v4i z4 = { 0, 0, 0, 0 };
  v8i z8 = { 0, 0, 0, 0, 0, 0, 0, 0 };
  __builtin_amdgcn_tensor_load_to_lds(g0, g1, z4, z4, z8, 0);
}

// ---- f32 -> f16 elementwise -----------------------------------------------
__global__ void k_cvt(const float* __restrict__ in, _Float16* __restrict__ out, int n) {
  int i = blockIdx.x * 256 + threadIdx.x;
  if (i < n) out[i] = (_Float16)in[i];
}

// ---- f32 W[k][n] -> f16 Wt[n][k] (transpose + convert) --------------------
__global__ void k_cvt_wT(const float* __restrict__ in, _Float16* __restrict__ out) {
  int i = blockIdx.x * 256 + threadIdx.x;   // over Cn*Cn; out index = n*Cn + k
  int nn = i >> 10, k = i & (Cn - 1);
  out[i] = (_Float16)in[k * Cn + nn];
}

// ---- QKV projection GEMM: x16(4096x1024) @ Wt -> Q,K row-major f16; V transposed
// grid (BT/64, Cn/128, 3), block 128 (4 waves). Weight tile (128x32) staged in
// LDS by TDM, double buffered; each wave DMAs its own 32-row slice.
__global__ __launch_bounds__(128) void k_gemm_qkv(
    const _Float16* __restrict__ x16,
    const _Float16* __restrict__ wqt, const _Float16* __restrict__ wkt,
    const _Float16* __restrict__ wvt,
    _Float16* __restrict__ q16, _Float16* __restrict__ k16,
    _Float16* __restrict__ vt16)
{
  __shared__ __align__(16) _Float16 bsm[2][128][32];   // 16 KB, double buffer

  const int z    = blockIdx.z;
  const _Float16* wt = (z == 0) ? wqt : (z == 1) ? wkt : wvt;
  const int lane = threadIdx.x & 31;
  const int wv   = threadIdx.x >> 5;
  const int lg   = lane >> 4, lm = lane & 15;
  const int row0 = (blockIdx.x * 4 + wv) * 16;
  const int col0 = blockIdx.y * 128;

  v8f acc[8] = {};
  const _Float16* arow  = x16 + (size_t)(row0 + lm) * Cn;
  const _Float16* wbase = wt + (size_t)(col0 + wv * 32) * Cn;

  tdm_load_tile32x32(wbase + 0,  (unsigned)(uintptr_t)&bsm[0][wv * 32][0]);
  tdm_load_tile32x32(wbase + 32, (unsigned)(uintptr_t)&bsm[1][wv * 32][0]);

  for (int i = 0; i < Cn / 32; i++) {
    const int k = i * 32, cur = i & 1;
    __builtin_amdgcn_s_wait_tensorcnt(1);   // our slice of buf[cur] is in LDS
    __syncthreads();                        // everyone's slice is in LDS

    v8h a0 = *(const v8h*)(arow + k + lg * 16);
    v8h a1 = *(const v8h*)(arow + k + lg * 16 + 8);
    v16h a = CAT16(a0, a1);

    // Batch all B fragments into distinct registers, then chain the WMMAs.
    v16h bfr[8];
#pragma unroll
    for (int nt = 0; nt < 8; nt++)
      bfr[nt] = *(const v16h*)&bsm[cur][nt * 16 + lm][lg * 16];
#pragma unroll
    for (int nt = 0; nt < 8; nt++)
      acc[nt] = WMMA_F16(a, bfr[nt], acc[nt]);

    __syncthreads();                        // done reading buf[cur]
    if (k + 64 < Cn)
      tdm_load_tile32x32(wbase + (k + 64), (unsigned)(uintptr_t)&bsm[cur][wv * 32][0]);
  }

  if (z < 2) {                       // Q / K: f16 row-major [B*T, C]
    _Float16* out = (z == 0) ? q16 : k16;
#pragma unroll
    for (int nt = 0; nt < 8; nt++) {
      int c = col0 + nt * 16 + lm;
#pragma unroll
      for (int r = 0; r < 8; r++)
        out[(size_t)(row0 + lg * 8 + r) * Cn + c] = (_Float16)acc[nt][r];
    }
  } else {                           // V: store transposed per head [B*H, d, T]
    int bidx = row0 >> 11;                 // / Tn
    int t0   = (row0 & (Tn - 1)) + lg * 8; // rows of C-tile = consecutive tokens
#pragma unroll
    for (int nt = 0; nt < 8; nt++) {
      int c = col0 + nt * 16 + lm;
      int h = c >> 6, dd = c & 63;
      v8h pk;
#pragma unroll
      for (int r = 0; r < 8; r++) pk[r] = (_Float16)acc[nt][r];
      *(v8h*)(vt16 + (size_t)((bidx * Hn + h) * Dn + dd) * Tn + t0) = pk;
    }
  }
}

// ---- Flash attention per (b,h) and 16-query tile --------------------------
// Computes S^T = K_tile(16x64) @ Q^T(64x16) so each lane owns one query.
// grid (B*H, T/16), block 32 (one wave)
__global__ __launch_bounds__(32) void k_attn(
    const _Float16* __restrict__ q16,   // [B*T, C] (d-contiguous per head)
    const _Float16* __restrict__ k16,   // [B*T, C]
    const _Float16* __restrict__ vt16,  // [B*H, d, T]
    _Float16* __restrict__ att16)       // [B*T, C]
{
  __shared__ __align__(16) _Float16 plds[16 * 32];   // P tile [query][key]

  const int bh = blockIdx.x;
  const int b  = bh >> 4, h = bh & 15;
  const int q0 = blockIdx.y * 16;
  const int lane = threadIdx.x;
  const int lg = lane >> 4, lm = lane & 15;
  const float scale = 0.03125f;        // C^-0.5 = 1/32

  // Q as B-matrix (d x 16 queries): lane lm = query column, K(=d) contiguous
  const _Float16* qrow = q16 + (size_t)(b * Tn + q0 + lm) * Cn + h * Dn;
  v16h bq0 = *(const v16h*)(qrow + lg * 16);        // d = 0..31
  v16h bq1 = *(const v16h*)(qrow + 32 + lg * 16);   // d = 32..63

  v8f o[4] = {};                       // O tiles: rows=query, cols=d (4x16)
  float mrun = -1e30f, lrun = 0.f;
  const int qg  = q0 + lm;             // query owned by this lane (softmax state)
  const int nkb = (q0 + 16 + 31) >> 5; // 32-key blocks up to causal frontier

  for (int jb = 0; jb < nkb; jb++) {
    const int kb0 = jb * 32;
    if (jb + 1 < nkb) {                // prefetch next key block (global_prefetch_b8)
      const _Float16* nk = k16 + (size_t)(b * Tn + kb0 + 32 + lm) * Cn + h * Dn;
      __builtin_prefetch(nk, 0, 1);
      __builtin_prefetch(nk + 16 * Cn, 0, 1);
    }
    v8f s[2] = {};                     // S^T tiles: rows=key, cols=query
#pragma unroll
    for (int sub = 0; sub < 2; sub++) {
      const _Float16* krow = k16 + (size_t)(b * Tn + kb0 + sub * 16 + lm) * Cn + h * Dn;
      v8h a0 = *(const v8h*)(krow + lg * 16);
      v8h a1 = *(const v8h*)(krow + lg * 16 + 8);
      s[sub] = WMMA_F16(CAT16(a0, a1), bq0, s[sub]);
      a0 = *(const v8h*)(krow + 32 + lg * 16);
      a1 = *(const v8h*)(krow + 40 + lg * 16);
      s[sub] = WMMA_F16(CAT16(a0, a1), bq1, s[sub]);
    }

    // scale + causal mask (branchless) + block max for this lane's query
    float mloc = -1e30f;
#pragma unroll
    for (int sub = 0; sub < 2; sub++)
#pragma unroll
      for (int r = 0; r < 8; r++) {
        int kgl = kb0 + sub * 16 + lg * 8 + r;     // key of this element
        float v = s[sub][r] * scale;
        v = (kgl <= qg) ? v : -1e30f;
        s[sub][r] = v;
        mloc = fmaxf(mloc, v);
      }
    float mnew = fmaxf(mrun, fmaxf(mloc, __shfl_xor(mloc, 16, 32)));
    float alpha = __expf(mrun - mnew);
    float lloc = 0.f;
#pragma unroll
    for (int sub = 0; sub < 2; sub++)
#pragma unroll
      for (int r = 0; r < 8; r++) {
        float p = __expf(s[sub][r] - mnew);
        s[sub][r] = p;
        lloc += p;
      }
    lloc += __shfl_xor(lloc, 16, 32);
    lrun = lrun * alpha + lloc;
    mrun = mnew;

    // rescale O rows (row m owned by lanes with m = r + lg*8; state lives at lane m)
#pragma unroll
    for (int r = 0; r < 8; r++) {
      float am = __shfl(alpha, lg * 8 + r, 32);
#pragma unroll
      for (int dt = 0; dt < 4; dt++) o[dt][r] *= am;
    }

    // P -> LDS [query][key]: each lane writes 8 consecutive keys (one b128)
#pragma unroll
    for (int sub = 0; sub < 2; sub++) {
      v8h ph;
#pragma unroll
      for (int r = 0; r < 8; r++) ph[r] = (_Float16)s[sub][r];
      *(v8h*)&plds[lm * 32 + sub * 16 + lg * 8] = ph;
    }
    // Re-read P in A-fragment layout (16 queries x 32 keys); same-wave DS is in-order
    v8h p0 = *(const v8h*)&plds[lm * 32 + lg * 16];
    v8h p1 = *(const v8h*)&plds[lm * 32 + lg * 16 + 8];
    v16h pa = CAT16(p0, p1);

    // O += P(16x32) @ V(32 keys x 64 d), V pre-transposed: keys contiguous
#pragma unroll
    for (int dt = 0; dt < 4; dt++) {
      v16h bv = *(const v16h*)(vt16 + (size_t)(bh * Dn + dt * 16 + lm) * Tn + kb0 + lg * 16);
      o[dt] = WMMA_F16(pa, bv, o[dt]);
    }
  }

  // Normalize and store f16 [B*T, C]
  float linv = 1.0f / lrun;
#pragma unroll
  for (int r = 0; r < 8; r++) {
    float li = __shfl(linv, lg * 8 + r, 32);
    size_t base = (size_t)(b * Tn + q0 + lg * 8 + r) * Cn + h * Dn + lm;
#pragma unroll
    for (int dt = 0; dt < 4; dt++)
      att16[base + dt * 16] = (_Float16)(o[dt][r] * li);
  }
}

// ---- Output projection: att16 @ Wpt + bp -> f32 out -----------------------
// grid (BT/64, Cn/128), block 128. Same TDM-staged weight tiles.
__global__ __launch_bounds__(128) void k_gemm_proj(
    const _Float16* __restrict__ a16, const _Float16* __restrict__ wpt,
    const float* __restrict__ bp, float* __restrict__ out)
{
  __shared__ __align__(16) _Float16 bsm[2][128][32];

  const int lane = threadIdx.x & 31;
  const int wv   = threadIdx.x >> 5;
  const int lg   = lane >> 4, lm = lane & 15;
  const int row0 = (blockIdx.x * 4 + wv) * 16;
  const int col0 = blockIdx.y * 128;

  v8f acc[8] = {};
  const _Float16* arow  = a16 + (size_t)(row0 + lm) * Cn;
  const _Float16* wbase = wpt + (size_t)(col0 + wv * 32) * Cn;

  tdm_load_tile32x32(wbase + 0,  (unsigned)(uintptr_t)&bsm[0][wv * 32][0]);
  tdm_load_tile32x32(wbase + 32, (unsigned)(uintptr_t)&bsm[1][wv * 32][0]);

  for (int i = 0; i < Cn / 32; i++) {
    const int k = i * 32, cur = i & 1;
    __builtin_amdgcn_s_wait_tensorcnt(1);
    __syncthreads();

    v8h a0 = *(const v8h*)(arow + k + lg * 16);
    v8h a1 = *(const v8h*)(arow + k + lg * 16 + 8);
    v16h a = CAT16(a0, a1);

    v16h bfr[8];
#pragma unroll
    for (int nt = 0; nt < 8; nt++)
      bfr[nt] = *(const v16h*)&bsm[cur][nt * 16 + lm][lg * 16];
#pragma unroll
    for (int nt = 0; nt < 8; nt++)
      acc[nt] = WMMA_F16(a, bfr[nt], acc[nt]);

    __syncthreads();
    if (k + 64 < Cn)
      tdm_load_tile32x32(wbase + (k + 64), (unsigned)(uintptr_t)&bsm[cur][wv * 32][0]);
  }

#pragma unroll
  for (int nt = 0; nt < 8; nt++) {
    int c = col0 + nt * 16 + lm;
    float bias = bp[c];
#pragma unroll
    for (int r = 0; r < 8; r++)
      out[(size_t)(row0 + lg * 8 + r) * Cn + c] = acc[nt][r] + bias;
  }
}

// ---------------------------------------------------------------------------
extern "C" void kernel_launch(void* const* d_in, const int* in_sizes, int n_in,
                              void* d_out, int out_size, void* d_ws, size_t ws_size,
                              hipStream_t stream) {
  const float* x  = (const float*)d_in[0];
  // d_in[1] = attention_mask: no-op in the reference
  const float* Wq = (const float*)d_in[2];
  const float* Wk = (const float*)d_in[3];
  const float* Wv = (const float*)d_in[4];
  const float* Wp = (const float*)d_in[5];
  const float* bp = (const float*)d_in[6];
  float* out = (float*)d_out;

  char* ws = (char*)d_ws;
  size_t off = 0;
  _Float16* x16  = (_Float16*)(ws + off); off += (size_t)BT * Cn * 2;
  _Float16* wqt  = (_Float16*)(ws + off); off += (size_t)Cn * Cn * 2;
  _Float16* wkt  = (_Float16*)(ws + off); off += (size_t)Cn * Cn * 2;
  _Float16* wvt  = (_Float16*)(ws + off); off += (size_t)Cn * Cn * 2;
  _Float16* wpt  = (_Float16*)(ws + off); off += (size_t)Cn * Cn * 2;
  _Float16* q16  = (_Float16*)(ws + off); off += (size_t)BT * Cn * 2;
  _Float16* k16  = (_Float16*)(ws + off); off += (size_t)BT * Cn * 2;
  _Float16* vt16 = (_Float16*)(ws + off); off += (size_t)BT * Cn * 2;
  _Float16* a16  = (_Float16*)(ws + off); off += (size_t)BT * Cn * 2;

  k_cvt<<<(BT * Cn) / 256, 256, 0, stream>>>(x, x16, BT * Cn);
  k_cvt_wT<<<(Cn * Cn) / 256, 256, 0, stream>>>(Wq, wqt);
  k_cvt_wT<<<(Cn * Cn) / 256, 256, 0, stream>>>(Wk, wkt);
  k_cvt_wT<<<(Cn * Cn) / 256, 256, 0, stream>>>(Wv, wvt);
  k_cvt_wT<<<(Cn * Cn) / 256, 256, 0, stream>>>(Wp, wpt);

  k_gemm_qkv<<<dim3(BT / 64, Cn / 128, 3), 128, 0, stream>>>(
      x16, wqt, wkt, wvt, q16, k16, vt16);

  k_attn<<<dim3(Bn * Hn, Tn / 16), 32, 0, stream>>>(q16, k16, vt16, a16);

  k_gemm_proj<<<dim3(BT / 64, Cn / 128), 128, 0, stream>>>(a16, wpt, bp, out);
}
